// SpectralGatingNetwork_25340307046819
// MI455X (gfx1250) — matrix operand order repaired and checked by reading, not compiled, not run on hardware
//
#include <hip/hip_runtime.h>
#include <hip/hip_bf16.h>

typedef _Float16 h16;
typedef __attribute__((ext_vector_type(16))) _Float16 v16h;
typedef __attribute__((ext_vector_type(8)))  _Float16 v8h;
typedef __attribute__((ext_vector_type(8)))  float    v8f;
typedef __attribute__((ext_vector_type(4)))  float    v4f;
typedef __attribute__((ext_vector_type(2)))  float    v2f;

constexpr int kB  = 64;
constexpr int kC  = 768;
constexpr int kH  = 64;
constexpr int kW  = 64;
constexpr int kWF = 33;   // rfft width
constexpr int kNP = 48;   // kw padded to 3 WMMA tiles
constexpr int LS  = 72;   // LDS stride in halves (bank-spread + 16B aligned)

// d_ws layout (in halves), all constants pre-laid-out in WMMA fragment order
constexpr int CRT_OFF = 0;      // B-frag: 3 ntiles * 2 ksteps * 32 lanes * 16
constexpr int CIT_OFF = 3072;
constexpr int FR_OFF  = 6144;   // A-frag: 4 mtiles * 2 ksteps * 32 * 16
constexpr int FI_OFF  = 10240;
constexpr int DR_OFF  = 14336;  // B-frag: 4 ntiles * 2 ksteps * 32 * 16
constexpr int DI_OFF  = 18432;
constexpr int CST_TOTAL = 22528; // halves (45,056 bytes)

// ---------------------------------------------------------------------------
// Init kernel: build twiddle matrices in fragment order.
//   CRT[k=w][n=kw] =  cos(2*pi*k*n/64)/64  (n<33 else 0)   -- stage-1 B
//   CIT[k=w][n=kw] = -sin(2*pi*k*n/64)/64                  -- stage-1 B
//   FR [m][k]      =  cos(2*pi*m*k/64)                     -- stage-3/5 A
//   FI [m][k]      = -sin(2*pi*m*k/64)
//   DR [k=kw][n=w] =  s_k*cos(2*pi*k*n/64)/64 (k<33 else 0)-- stage-6 B
//   DI [k=kw][n=w] = -s_k*sin(2*pi*k*n/64)/64
// Angle only depends on (k*n) mod 64 -> exact small arguments.
// ---------------------------------------------------------------------------
__global__ void sgn_twiddle_init(h16* __restrict__ cst) {
  int idx = blockIdx.x * 256 + threadIdx.x;
  if (idx >= CST_TOTAL) return;
  const float ANG = 0.09817477042468103f;  // pi/32
  float val = 0.0f;

  if (idx < FR_OFF) {                       // CRT / CIT  (B-frag, N=48)
    int isI = (idx >= CIT_OFF);
    int r = idx - (isI ? CIT_OFF : CRT_OFF);
    int j = r & 15, lane = (r >> 4) & 31, bi = r >> 9;
    int kstep = bi & 1, ntile = bi >> 1;
    int k = kstep * 32 + ((lane >> 4) << 3) + ((j < 8) ? j : j + 8);   // w
    int n = ntile * 16 + (lane & 15);              // kw
    if (n < kWF) {
      float s, c; __sincosf((float)((k * n) & 63) * ANG, &s, &c);
      val = (isI ? -s : c) * (1.0f / 64.0f);
    }
  } else if (idx < DR_OFF) {                // FR / FI  (A-frag)
    int r = idx - FR_OFF;
    int isI = (r >= 4096); r -= isI ? 4096 : 0;
    int j = r & 15, lane = (r >> 4) & 31, bi = r >> 9;
    int kstep = bi & 1, mtile = bi >> 1;
    int m  = mtile * 16 + (lane & 15);
    int kk = kstep * 32 + ((lane >> 4) << 4) + j;
    float s, c; __sincosf((float)((m * kk) & 63) * ANG, &s, &c);
    val = isI ? -s : c;
  } else {                                  // DR / DI  (B-frag, N=64)
    int r = idx - DR_OFF;
    int isI = (r >= 4096); r -= isI ? 4096 : 0;
    int j = r & 15, lane = (r >> 4) & 31, bi = r >> 9;
    int kstep = bi & 1, ntile = bi >> 1;
    int k = kstep * 32 + ((lane >> 4) << 3) + ((j < 8) ? j : j + 8);   // kw
    int n = ntile * 16 + (lane & 15);              // w
    if (k < kWF) {
      float sk = (k == 0 || k == 32) ? 1.0f : 2.0f;
      float s, c; __sincosf((float)((k * n) & 63) * ANG, &s, &c);
      val = (isI ? -s : c) * sk * (1.0f / 64.0f);
    }
  }
  cst[idx] = (h16)val;
}

// ---------------------------------------------------------------------------
// Fragment helpers
// ---------------------------------------------------------------------------
__device__ __forceinline__ v8f wmma_f16(v16h a, v16h b, v8f c) {
  return __builtin_amdgcn_wmma_f32_16x16x32_f16(false, a, false, b,
                                                (short)0, c, false, false);
}

// 16 contiguous halves (B-fragments / pre-fragmented constants), 16B aligned
__device__ __forceinline__ v16h ld_contig16(const h16* p) {
  union { v16h v; v8h h[2]; } u;
  u.h[0] = *(const v8h*)p;
  u.h[1] = *(const v8h*)(p + 8);
  return u.v;
}

// A-fragment from a half row in LDS (row-major, lane already picked row):
// e[0..7] = row[kb..kb+7], e[8..15] = row[kb+16..kb+23]
__device__ __forceinline__ v16h ldA_half(const h16* row, int kb) {
  union { v16h v; v8h h[2]; } u;
  u.h[0] = *(const v8h*)(row + kb);
  u.h[1] = *(const v8h*)(row + kb + 16);
  return u.v;
}

// A-fragment directly from a global f32 row, converting to f16
__device__ __forceinline__ v16h ldA_f32(const float* row, int kb) {
  union { v16h v; h16 e[16]; } u;
  v4f f0 = *(const v4f*)(row + kb);
  v4f f1 = *(const v4f*)(row + kb + 4);
  v4f f2 = *(const v4f*)(row + kb + 16);
  v4f f3 = *(const v4f*)(row + kb + 20);
#pragma unroll
  for (int i = 0; i < 4; ++i) {
    u.e[i]      = (h16)f0[i];
    u.e[4 + i]  = (h16)f1[i];
    u.e[8 + i]  = (h16)f2[i];
    u.e[12 + i] = (h16)f3[i];
  }
  return u.v;
}

// ---------------------------------------------------------------------------
// Main kernel: one 64x64 image per workgroup, 4 waves (wave32), wave w owns
// row-tile w of every stage's output.
// ---------------------------------------------------------------------------
__global__ __launch_bounds__(128) void sgn_main(
    const float* __restrict__ x,     // (B, C, H, W) f32
    const float* __restrict__ wgt,   // (H, WF, C, 2) f32
    const h16*  __restrict__ cst,    // pre-fragmented twiddles
    float* __restrict__ out)         // (B, C, H, W) f32
{
  // Inter-stage tiles, N-major (row = N index of next stage's B operand)
  __shared__ alignas(16) h16 XRt[kNP * LS];   // rfft row-stage real
  __shared__ alignas(16) h16 XIt[kNP * LS];   // rfft row-stage imag
  __shared__ alignas(16) h16 VRt[kNP * LS];   // gated spectrum real
  __shared__ alignas(16) h16 VIt[kNP * LS];   // gated spectrum imag
  // Stage-5 output, A-layout (row-major, K = kw padded to 64)
  __shared__ alignas(16) h16 ZRa[kH * LS];
  __shared__ alignas(16) h16 ZIa[kH * LS];

  const int tid   = threadIdx.x;
  const int lane  = tid & 31;
  const int mtile = tid >> 5;              // wave id = row tile
  const int hi16  = lane >> 4;             // 0 or 1
  const int lo16  = lane & 15;

  const int c = blockIdx.x % kC;
  const int b = blockIdx.x / kC;
  const float* xim  = x   + ((size_t)b * kC + c) * (kH * kW);
  float*       yim  = out + ((size_t)b * kC + c) * (kH * kW);

  // Zero-pad ZRa/ZIa columns 48..63 (K padding for stage 6)
  for (int i = tid; i < kH * 16; i += 128) {
    int r = i >> 4, cc = 48 + (i & 15);
    ZRa[r * LS + cc] = (h16)0.0f;
    ZIa[r * LS + cc] = (h16)0.0f;
  }

  // ------------------- Stage 1: XR = X*CRT, XI = X*CIT -------------------
  {
    const int m = mtile * 16 + lo16;
    const float* xrow = xim + m * kW;
    v16h a0 = ldA_f32(xrow, 0 * 32 + hi16 * 8);
    v16h a1 = ldA_f32(xrow, 1 * 32 + hi16 * 8);
#pragma unroll
    for (int n = 0; n < 3; ++n) {
      v16h cr0 = ld_contig16(cst + CRT_OFF + ((n * 2 + 0) * 32 + lane) * 16);
      v16h cr1 = ld_contig16(cst + CRT_OFF + ((n * 2 + 1) * 32 + lane) * 16);
      v16h ci0 = ld_contig16(cst + CIT_OFF + ((n * 2 + 0) * 32 + lane) * 16);
      v16h ci1 = ld_contig16(cst + CIT_OFF + ((n * 2 + 1) * 32 + lane) * 16);
      v8f aR = {0.f,0.f,0.f,0.f,0.f,0.f,0.f,0.f};
      v8f aI = {0.f,0.f,0.f,0.f,0.f,0.f,0.f,0.f};
      aR = wmma_f16(a0, cr0, aR); aR = wmma_f16(a1, cr1, aR);
      aI = wmma_f16(a0, ci0, aI); aI = wmma_f16(a1, ci1, aI);
#pragma unroll
      for (int v = 0; v < 8; ++v) {
        int gm = mtile * 16 + v + (hi16 << 3);   // h
        int gn = n * 16 + lo16;                  // kw
        XRt[gn * LS + gm] = (h16)aR[v];
        XIt[gn * LS + gm] = (h16)aI[v];
      }
    }
  }
  __syncthreads();

  // A operands for stages 3 & 5: FR/FI rows of this wave's kh/h tile
  v16h fr0 = ld_contig16(cst + FR_OFF + ((mtile * 2 + 0) * 32 + lane) * 16);
  v16h fr1 = ld_contig16(cst + FR_OFF + ((mtile * 2 + 1) * 32 + lane) * 16);
  v16h fi0 = ld_contig16(cst + FI_OFF + ((mtile * 2 + 0) * 32 + lane) * 16);
  v16h fi1 = ld_contig16(cst + FI_OFF + ((mtile * 2 + 1) * 32 + lane) * 16);

  // --------- Stage 3+4: U = F * X1 (complex), then elementwise gate -------
#pragma unroll
  for (int n = 0; n < 3; ++n) {
    v16h xr0 = ld_contig16(&XRt[(n * 16 + lo16) * LS + (0 * 32 + hi16 * 16)]);
    v16h xr1 = ld_contig16(&XRt[(n * 16 + lo16) * LS + (1 * 32 + hi16 * 16)]);
    v16h xi0 = ld_contig16(&XIt[(n * 16 + lo16) * LS + (0 * 32 + hi16 * 16)]);
    v16h xi1 = ld_contig16(&XIt[(n * 16 + lo16) * LS + (1 * 32 + hi16 * 16)]);
    v8f pr = {0.f,0.f,0.f,0.f,0.f,0.f,0.f,0.f};  // FR*XR
    v8f qr = {0.f,0.f,0.f,0.f,0.f,0.f,0.f,0.f};  // FI*XI
    v8f pi = {0.f,0.f,0.f,0.f,0.f,0.f,0.f,0.f};  // FR*XI
    v8f qi = {0.f,0.f,0.f,0.f,0.f,0.f,0.f,0.f};  // FI*XR
    pr = wmma_f16(fr0, xr0, pr); pr = wmma_f16(fr1, xr1, pr);
    qr = wmma_f16(fi0, xi0, qr); qr = wmma_f16(fi1, xi1, qr);
    pi = wmma_f16(fr0, xi0, pi); pi = wmma_f16(fr1, xi1, pi);
    qi = wmma_f16(fi0, xr0, qi); qi = wmma_f16(fi1, xr1, qi);
#pragma unroll
    for (int v = 0; v < 8; ++v) {
      float UR = pr[v] - qr[v];
      float UI = pi[v] + qi[v];
      int kh = mtile * 16 + v + (hi16 << 3);
      int kw = n * 16 + lo16;
      float gr = 0.f, gi = 0.f;
      if (kw < kWF) {
        v2f g = *(const v2f*)(wgt + (((size_t)kh * kWF + kw) * kC + c) * 2);
        gr = g[0]; gi = g[1];
      }
      float VR = UR * gr - UI * gi;
      float VI = UR * gi + UI * gr;
      VRt[kw * LS + kh] = (h16)VR;
      VIt[kw * LS + kh] = (h16)VI;
    }
  }
  __syncthreads();

  // ------------- Stage 5: Z = conj(F) * V  ->  ZRa/ZIa (A-layout) --------
#pragma unroll
  for (int n = 0; n < 3; ++n) {
    v16h vr0 = ld_contig16(&VRt[(n * 16 + lo16) * LS + (0 * 32 + hi16 * 16)]);
    v16h vr1 = ld_contig16(&VRt[(n * 16 + lo16) * LS + (1 * 32 + hi16 * 16)]);
    v16h vi0 = ld_contig16(&VIt[(n * 16 + lo16) * LS + (0 * 32 + hi16 * 16)]);
    v16h vi1 = ld_contig16(&VIt[(n * 16 + lo16) * LS + (1 * 32 + hi16 * 16)]);
    v8f zr = {0.f,0.f,0.f,0.f,0.f,0.f,0.f,0.f};  // FR*VR + FI*VI
    v8f pz = {0.f,0.f,0.f,0.f,0.f,0.f,0.f,0.f};  // FR*VI
    v8f qz = {0.f,0.f,0.f,0.f,0.f,0.f,0.f,0.f};  // FI*VR
    zr = wmma_f16(fr0, vr0, zr); zr = wmma_f16(fr1, vr1, zr);
    zr = wmma_f16(fi0, vi0, zr); zr = wmma_f16(fi1, vi1, zr);
    pz = wmma_f16(fr0, vi0, pz); pz = wmma_f16(fr1, vi1, pz);
    qz = wmma_f16(fi0, vr0, qz); qz = wmma_f16(fi1, vr1, qz);
#pragma unroll
    for (int v = 0; v < 8; ++v) {
      int gm = mtile * 16 + v + (hi16 << 3);   // h
      int gn = n * 16 + lo16;                  // kw
      ZRa[gm * LS + gn] = (h16)zr[v];
      ZIa[gm * LS + gn] = (h16)(pz[v] - qz[v]);
    }
  }
  __syncthreads();

  // ----------- Stage 6: Y = ZR*DR + ZI*DI  -> global output --------------
  {
    const h16* zrrow = &ZRa[(mtile * 16 + lo16) * LS];
    const h16* zirow = &ZIa[(mtile * 16 + lo16) * LS];
    v16h zr0 = ldA_half(zrrow, 0 * 32 + hi16 * 8);
    v16h zr1 = ldA_half(zrrow, 1 * 32 + hi16 * 8);
    v16h zi0 = ldA_half(zirow, 0 * 32 + hi16 * 8);
    v16h zi1 = ldA_half(zirow, 1 * 32 + hi16 * 8);
#pragma unroll
    for (int n = 0; n < 4; ++n) {
      v16h dr0 = ld_contig16(cst + DR_OFF + ((n * 2 + 0) * 32 + lane) * 16);
      v16h dr1 = ld_contig16(cst + DR_OFF + ((n * 2 + 1) * 32 + lane) * 16);
      v16h di0 = ld_contig16(cst + DI_OFF + ((n * 2 + 0) * 32 + lane) * 16);
      v16h di1 = ld_contig16(cst + DI_OFF + ((n * 2 + 1) * 32 + lane) * 16);
      v8f acc = {0.f,0.f,0.f,0.f,0.f,0.f,0.f,0.f};
      acc = wmma_f16(zr0, dr0, acc); acc = wmma_f16(zr1, dr1, acc);
      acc = wmma_f16(zi0, di0, acc); acc = wmma_f16(zi1, di1, acc);
#pragma unroll
      for (int v = 0; v < 8; ++v) {
        int h = mtile * 16 + v + (hi16 << 3);
        int w = n * 16 + lo16;
        yim[h * kW + w] = acc[v];
      }
    }
  }
}

extern "C" void kernel_launch(void* const* d_in, const int* in_sizes, int n_in,
                              void* d_out, int out_size, void* d_ws, size_t ws_size,
                              hipStream_t stream) {
  (void)in_sizes; (void)n_in; (void)out_size;
  if (ws_size < (size_t)CST_TOTAL * sizeof(h16)) return;

  const float* x   = (const float*)d_in[0];
  const float* wgt = (const float*)d_in[1];
  float*       out = (float*)d_out;
  h16*         cst = (h16*)d_ws;

  // Build twiddle constants in fragment order (deterministic each call)
  sgn_twiddle_init<<<(CST_TOTAL + 255) / 256, 256, 0, stream>>>(cst);

  // One workgroup (4 waves) per (b, c) image
  sgn_main<<<kB * kC, 128, 0, stream>>>(x, wgt, cst, out);
}
